// AttentionLayer_56753697849685
// MI455X (gfx1250) — compile-verified, matrix operand order checked by reading, not run on hardware
//
#include <hip/hip_runtime.h>
#include <hip/hip_bf16.h>

#define EMB 512
#define FF  64
#define BB  8
#define SS  2048
#define TOT (BB * SS)   // 16384 rows

typedef __attribute__((ext_vector_type(16))) _Float16 v16h;
typedef __attribute__((ext_vector_type(8)))  float    v8f;

#if __has_builtin(__builtin_amdgcn_tensor_load_to_lds)
#define HAS_TDM 1
#else
#define HAS_TDM 0
#endif

union V16HU { v16h v; uint4 q[2]; _Float16 h[16]; };

// ---------------- WMMA fragment loaders (ISA 7.12.2 layouts) ----------------
// A (16x32 f16, MxK): lane<16 holds row M=lane, K={k0..k0+7, k0+16..k0+23};
// lane>=16 holds row M=lane-16, K={k0+8..k0+15, k0+24..k0+31}.
// B (32x16, KxN) from a row-major [N][K] source has the mirrored layout:
// lane holds column N=lane%16 -> same gather with "row = N".
__device__ __forceinline__ v16h frag_ld_f16(const _Float16* __restrict__ p, int ld) {
  const int lane = threadIdx.x & 31;
  const int grp  = lane >> 4;
  const int r    = lane & 15;
  const _Float16* row = p + r * ld;
  V16HU u;
  u.q[0] = *(const uint4*)(row + (grp ? 8 : 0));
  u.q[1] = *(const uint4*)(row + (grp ? 24 : 16));
  return u.v;
}

__device__ __forceinline__ v16h frag_ld_f32cvt(const float* __restrict__ p, int ld) {
  const int lane = threadIdx.x & 31;
  const int grp  = lane >> 4;
  const int r    = lane & 15;
  const float* row = p + r * ld;
  const int ka = grp ? 8 : 0;
  const int kb = grp ? 24 : 16;
  v16h a;
#pragma unroll
  for (int i = 0; i < 8; ++i) {
    a[i]     = (_Float16)row[ka + i];
    a[i + 8] = (_Float16)row[kb + i];
  }
  return a;
}

__device__ __forceinline__ v8f wmma_f16(v16h a, v16h b, v8f c) {
  return __builtin_amdgcn_wmma_f32_16x16x32_f16(false, a, false, b, (short)0, c, false, false);
}

__device__ __forceinline__ v8f v8f_zero() {
  v8f c;
#pragma unroll
  for (int r = 0; r < 8; ++r) c[r] = 0.0f;
  return c;
}

// ---------------- TDM: 2D f16 tile DMA global -> LDS (ISA ch.8 D#) ----------
// Groups packed per ISA 8.3/8.4: g0 = {count=1, lds_addr, global_addr, type=2},
// g1 = {data_size=2B, tensor_dim0/1, tile_dim0/1, tensor_dim0_stride}.
// 6-arg builtin (clang-23 lane): (u32x4, i32x8, i32x4, i32x4, i32x8, i32 cpol).
__device__ __forceinline__ void tdm_load_2d_f16(const void* gptr, unsigned lds_addr,
                                                unsigned cols, unsigned rows,
                                                unsigned ld, unsigned tensor_rows) {
#if HAS_TDM
  typedef __attribute__((ext_vector_type(4))) unsigned int u32x4;
  typedef __attribute__((ext_vector_type(8))) int i32x8;
  typedef __attribute__((ext_vector_type(4))) int i32x4;
  unsigned long long ga = (unsigned long long)(uintptr_t)gptr;
  u32x4 g0;
  g0[0] = 1u;                                                  // count=1 (user D#)
  g0[1] = lds_addr;                                            // lds_addr bytes
  g0[2] = (unsigned)(ga & 0xffffffffu);                        // global_addr[31:0]
  g0[3] = (unsigned)((ga >> 32) & 0x01ffffffu) | (2u << 30);   // addr[56:32] | type=2
  i32x8 g1;
  g1[0] = (int)(1u << 16);                                     // data_size=1 -> 2 bytes
  g1[1] = (int)((ld & 0xffffu) << 16);                         // tensor_dim0 lo16 @bit48
  g1[2] = (int)(((ld >> 16) & 0xffffu) | ((tensor_rows & 0xffffu) << 16)); // dim0 hi | dim1 lo
  g1[3] = (int)(((tensor_rows >> 16) & 0xffffu) | ((cols & 0xffffu) << 16)); // dim1 hi | tile_dim0
  g1[4] = (int)(rows & 0xffffu);                               // tile_dim1 (tile_dim2=0)
  g1[5] = (int)ld;                                             // tensor_dim0_stride lo32
  g1[6] = 0;
  g1[7] = 0;
  i32x4 gz4; gz4[0] = 0; gz4[1] = 0; gz4[2] = 0; gz4[3] = 0;
  i32x8 gz8;
#pragma unroll
  for (int i = 0; i < 8; ++i) gz8[i] = 0;
  __builtin_amdgcn_tensor_load_to_lds(g0, g1, gz4, gz4, gz8, 0);
#else
  (void)gptr; (void)lds_addr; (void)cols; (void)rows; (void)ld; (void)tensor_rows;
#endif
}

// ---------------- kernel 1: f32 -> f16 weight convert ----------------
__global__ void attn56_cvt_kernel(const float* __restrict__ src, _Float16* __restrict__ dst, int n) {
  int i = blockIdx.x * 256 + threadIdx.x;
  if (i < n) dst[i] = (_Float16)src[i];
}

// ---------------- kernel 2: QKV projections ----------------
// y = X @ W^T + b ; q scaled by 1/sqrt(512); v stored transposed [E][TOT]
__global__ __launch_bounds__(128) void attn56_qkv_kernel(
    const float* __restrict__ query, const float* __restrict__ key, const float* __restrict__ value,
    const _Float16* __restrict__ Wq16, const _Float16* __restrict__ Wk16, const _Float16* __restrict__ Wv16,
    const float* __restrict__ bq, const float* __restrict__ bk, const float* __restrict__ bv,
    _Float16* __restrict__ q16, _Float16* __restrict__ k16, _Float16* __restrict__ vt16)
{
  const int MT = TOT / 16, NT = EMB / 16;            // 1024 x 32 tiles
  const int wid = blockIdx.x * 4 + (threadIdx.x >> 5);
  const int sel = wid / (MT * NT);
  int rem = wid - sel * (MT * NT);
  const int m0 = (rem / NT) * 16;
  const int n0 = (rem % NT) * 16;
  const int lane = threadIdx.x & 31;
  const int grp = lane >> 4, coln = lane & 15;

  const float* X; const _Float16* W; const float* bias;
  if (sel == 0)      { X = query; W = Wq16; bias = bq; }
  else if (sel == 1) { X = key;   W = Wk16; bias = bk; }
  else               { X = value; W = Wv16; bias = bv; }

  const float bval = bias[n0 + coln];

  // two independent accumulator chains for WMMA ILP
  v8f c0 = v8f_zero(), c1 = v8f_zero();
  for (int k0 = 0; k0 < EMB; k0 += 64) {
    v16h a0 = frag_ld_f32cvt(X + m0 * EMB + k0, EMB);
    v16h a1 = frag_ld_f32cvt(X + m0 * EMB + k0 + 32, EMB);
    v16h b0 = frag_ld_f16(W + n0 * EMB + k0, EMB);
    v16h b1 = frag_ld_f16(W + n0 * EMB + k0 + 32, EMB);
    c0 = wmma_f16(a0, b0, c0);
    c1 = wmma_f16(a1, b1, c1);
  }
  v8f c = c0 + c1;

  if (sel == 0) {
#pragma unroll
    for (int r = 0; r < 8; ++r) {
      int m = m0 + r + 8 * grp;
      q16[m * EMB + n0 + coln] = (_Float16)((c[r] + bval) * 0.04419417382415922f); // 1/sqrt(512)
    }
  } else if (sel == 1) {
#pragma unroll
    for (int r = 0; r < 8; ++r) {
      int m = m0 + r + 8 * grp;
      k16[m * EMB + n0 + coln] = (_Float16)(c[r] + bval);
    }
  } else {
    _Float16 tmp[8];
#pragma unroll
    for (int r = 0; r < 8; ++r) tmp[r] = (_Float16)(c[r] + bval);
    // vt[f][m]: 8 consecutive rows m -> one 16B store per lane
    *(uint4*)(vt16 + (n0 + coln) * TOT + m0 + 8 * grp) = *(uint4*)tmp;
  }
}

// ---------------- kernel 3: attention + residual + LN1 ----------------
// One workgroup (4 waves) per (batch, 16-query-row tile).
// LDS: scores f16 [16][2048] (reused as f32 out tile), P f16 [16][2048],
// scratch, Q tile staging (TDM). 148 KB/block -> 2 blocks per 320KB WGP.
#define SC16_BYTES (16 * SS * 2)
#define P16_OFF    SC16_BYTES
#define RED_OFF    (P16_OFF + 16 * SS * 2)
#define STAT_OFF   (RED_OFF + 256 * 4)
#define QS_OFF     (STAT_OFF + 32 * 4)
#define SMEM_BYTES (QS_OFF + 16 * EMB * 2)

#define MASKED_F16 ((_Float16)(-65504.0f))

__global__ __launch_bounds__(128) void attn56_attn_kernel(
    const _Float16* __restrict__ q16, const _Float16* __restrict__ k16,
    const _Float16* __restrict__ vt16, const int* __restrict__ mask,
    const float* __restrict__ query, const float* __restrict__ g1, const float* __restrict__ be1,
    float* __restrict__ out1f32, _Float16* __restrict__ out1f16)
{
  extern __shared__ __align__(16) char smem[];
  _Float16* sc16 = (_Float16*)smem;
  _Float16* p16  = (_Float16*)(smem + P16_OFF);
  float*    red  = (float*)(smem + RED_OFF);   // 256 floats
  float*    stat = (float*)(smem + STAT_OFF);  // 32 floats
  _Float16* qs   = (_Float16*)(smem + QS_OFF); // 16x512 staged Q tile
  float*    outs = (float*)smem;               // alias sc16 (32KB of 64KB)

  const int b    = blockIdx.x >> 7;     // 128 q-tiles per batch
  const int qt   = blockIdx.x & 127;
  const int row0 = b * SS + qt * 16;    // global row base
  const int wv   = threadIdx.x >> 5;
  const int lane = threadIdx.x & 31;
  const int grp  = lane >> 4, coln = lane & 15;
  const int tid  = threadIdx.x;

  // ---- stage Q tile (16x512 f16) to LDS via Tensor Data Mover ----
#if HAS_TDM
  if (wv == 0) {
    tdm_load_2d_f16(q16 + row0 * EMB, (unsigned)(uintptr_t)qs, EMB, 16, EMB, TOT);
    __builtin_amdgcn_s_wait_tensorcnt(0);
  }
#else
  for (int i = tid; i < 16 * EMB / 8; i += 128)
    ((uint4*)qs)[i] = ((const uint4*)(q16 + row0 * EMB))[i];
#endif
  __syncthreads();

  // ---- preload Q tile into 16 registered A-fragments (from LDS) ----
  v16h qfrag[16];
#pragma unroll
  for (int kk = 0; kk < 16; ++kk)
    qfrag[kk] = frag_ld_f16(qs + kk * 32, EMB);

  // ---- phase 1: scores = q @ k^T (scale pre-folded), mask, -> LDS f16 ----
  for (int kt = wv; kt < SS / 16; kt += 4) {
    const _Float16* kbase = k16 + (b * SS + kt * 16) * EMB;
    if (kt + 4 < SS / 16) {
      // prefetch next key tile + mask tile (global_prefetch_b8)
      __builtin_prefetch(k16 + (b * SS + (kt + 4) * 16) * EMB, 0, 1);
      __builtin_prefetch(mask + row0 * SS + (kt + 4) * 16, 0, 1);
    }
    v8f c0 = v8f_zero(), c1 = v8f_zero(), c2 = v8f_zero(), c3 = v8f_zero();
#pragma unroll
    for (int kk = 0; kk < 16; kk += 4) {
      v16h b0 = frag_ld_f16(kbase + (kk + 0) * 32, EMB);
      v16h b1 = frag_ld_f16(kbase + (kk + 1) * 32, EMB);
      v16h b2 = frag_ld_f16(kbase + (kk + 2) * 32, EMB);
      v16h b3 = frag_ld_f16(kbase + (kk + 3) * 32, EMB);
      c0 = wmma_f16(qfrag[kk + 0], b0, c0);
      c1 = wmma_f16(qfrag[kk + 1], b1, c1);
      c2 = wmma_f16(qfrag[kk + 2], b2, c2);
      c3 = wmma_f16(qfrag[kk + 3], b3, c3);
    }
    v8f c = (c0 + c1) + (c2 + c3);
    const int kc = kt * 16 + coln;
#pragma unroll
    for (int r = 0; r < 8; ++r) {
      int m = r + 8 * grp;
      _Float16 v = (_Float16)c[r];
      if (mask[(row0 + m) * SS + kc] == 0) v = MASKED_F16;
      sc16[m * SS + kc] = v;
    }
  }
  __syncthreads();

  // ---- phase 2: softmax over LDS rows (16 rows x 8 threads each) ----
  const int srow = tid >> 3, seg = tid & 7;
  {
    float mx = -3.4e38f;
    for (int cix = seg * 256; cix < seg * 256 + 256; ++cix)
      mx = fmaxf(mx, (float)sc16[srow * SS + cix]);
    red[srow * 8 + seg] = mx;
  }
  __syncthreads();
  if (tid < 16) {
    float mx = red[tid * 8];
#pragma unroll
    for (int i = 1; i < 8; ++i) mx = fmaxf(mx, red[tid * 8 + i]);
    stat[tid] = mx;
  }
  __syncthreads();
  {
    float rmax = stat[srow], s = 0.f;
    for (int cix = seg * 256; cix < seg * 256 + 256; ++cix) {
      float e = __expf((float)sc16[srow * SS + cix] - rmax);
      p16[srow * SS + cix] = (_Float16)e;   // unnormalized; scale folded into P@V output
      s += e;
    }
    red[srow * 8 + seg] = s;
  }
  __syncthreads();
  if (tid < 16) {
    float s = 0.f;
#pragma unroll
    for (int i = 0; i < 8; ++i) s += red[tid * 8 + i];
    stat[16 + tid] = 1.0f / s;
  }
  __syncthreads();

  // ---- phase 3: attn = P @ V (A from LDS, 4 WMMA chains), + residual ----
  for (int ct = wv; ct < EMB / 16; ct += 4) {
    const int e0 = ct * 16;
    const _Float16* vbase = vt16 + e0 * TOT + b * SS;
    v8f c0 = v8f_zero(), c1 = v8f_zero(), c2 = v8f_zero(), c3 = v8f_zero();
    for (int k0 = 0; k0 < SS; k0 += 128) {
      v16h a0 = frag_ld_f16(p16 + k0, SS);
      v16h a1 = frag_ld_f16(p16 + k0 + 32, SS);
      v16h a2 = frag_ld_f16(p16 + k0 + 64, SS);
      v16h a3 = frag_ld_f16(p16 + k0 + 96, SS);
      v16h b0 = frag_ld_f16(vbase + k0, TOT);
      v16h b1 = frag_ld_f16(vbase + k0 + 32, TOT);
      v16h b2 = frag_ld_f16(vbase + k0 + 64, TOT);
      v16h b3 = frag_ld_f16(vbase + k0 + 96, TOT);
      c0 = wmma_f16(a0, b0, c0);
      c1 = wmma_f16(a1, b1, c1);
      c2 = wmma_f16(a2, b2, c2);
      c3 = wmma_f16(a3, b3, c3);
    }
    v8f c = (c0 + c1) + (c2 + c3);
#pragma unroll
    for (int r = 0; r < 8; ++r) {
      int m = r + 8 * grp;
      float o = c[r] * stat[16 + m] + query[(row0 + m) * EMB + e0 + coln];
      outs[m * EMB + e0 + coln] = o;
    }
  }
  __syncthreads();

  // ---- phase 4: LayerNorm1 (16 rows x 8 threads x 64 cols) ----
  {
    float s = 0.f, ss = 0.f;
    for (int cix = seg * 64; cix < seg * 64 + 64; ++cix) {
      float v = outs[srow * EMB + cix];
      s += v; ss += v * v;
    }
    red[srow * 8 + seg] = s;
    red[128 + srow * 8 + seg] = ss;
  }
  __syncthreads();
  if (tid < 16) {
    float s = 0.f, ss = 0.f;
#pragma unroll
    for (int i = 0; i < 8; ++i) { s += red[tid * 8 + i]; ss += red[128 + tid * 8 + i]; }
    float mu = s * (1.0f / EMB);
    float var = ss * (1.0f / EMB) - mu * mu;
    stat[tid] = mu;
    stat[16 + tid] = rsqrtf(var + 1e-5f);
  }
  __syncthreads();
  {
    float mu = stat[srow], rs = stat[16 + srow];
    for (int cix = seg * 64; cix < seg * 64 + 64; ++cix) {
      float v = (outs[srow * EMB + cix] - mu) * rs * g1[cix] + be1[cix];
      out1f32[(row0 + srow) * EMB + cix] = v;
      out1f16[(row0 + srow) * EMB + cix] = (_Float16)v;
    }
  }
}

// ---------------- kernel 4: FFN + residual + LN2 ----------------
__global__ __launch_bounds__(128) void attn56_ffn_kernel(
    const float* __restrict__ out1f32, const _Float16* __restrict__ out1f16,
    const _Float16* __restrict__ W116, const _Float16* __restrict__ W216,
    const float* __restrict__ b1, const float* __restrict__ b2,
    const float* __restrict__ g2, const float* __restrict__ be2,
    float* __restrict__ out)
{
  __shared__ __align__(16) _Float16 xs[16 * EMB];   // TDM-staged A tile
  __shared__ __align__(16) _Float16 h16[16 * FF];
  __shared__ __align__(16) float outs[16 * EMB];
  __shared__ float red[256];
  __shared__ float stat[32];

  const int row0 = blockIdx.x * 16;
  const int wv = threadIdx.x >> 5, lane = threadIdx.x & 31;
  const int grp = lane >> 4, coln = lane & 15, tid = threadIdx.x;

  // ---- stage out1 tile (16x512 f16) to LDS via TDM ----
#if HAS_TDM
  if (wv == 0) {
    tdm_load_2d_f16(out1f16 + row0 * EMB, (unsigned)(uintptr_t)xs, EMB, 16, EMB, TOT);
    __builtin_amdgcn_s_wait_tensorcnt(0);
  }
#else
  for (int i = tid; i < 16 * EMB / 8; i += 128)
    ((uint4*)xs)[i] = ((const uint4*)(out1f16 + row0 * EMB))[i];
#endif
  __syncthreads();

  // phase A: h = relu(out1 @ W1^T + b1), wave wv -> cols [16wv,16wv+16)
  {
    const float bv = b1[wv * 16 + coln];
    v8f c0 = v8f_zero(), c1 = v8f_zero();
    for (int k0 = 0; k0 < EMB; k0 += 64) {
      v16h a0 = frag_ld_f16(xs + k0, EMB);
      v16h a1 = frag_ld_f16(xs + k0 + 32, EMB);
      v16h b0 = frag_ld_f16(W116 + (wv * 16) * EMB + k0, EMB);
      v16h b1f = frag_ld_f16(W116 + (wv * 16) * EMB + k0 + 32, EMB);
      c0 = wmma_f16(a0, b0, c0);
      c1 = wmma_f16(a1, b1f, c1);
    }
    v8f c = c0 + c1;
#pragma unroll
    for (int r = 0; r < 8; ++r) {
      int m = r + 8 * grp;
      h16[m * FF + wv * 16 + coln] = (_Float16)fmaxf(c[r] + bv, 0.0f);
    }
  }
  __syncthreads();

  // phase B: ff = h @ W2^T + b2 + out1 (residual)
  for (int ct = wv; ct < EMB / 16; ct += 4) {
    const int e0 = ct * 16;
    const float bv = b2[e0 + coln];
    v8f c0 = v8f_zero(), c1 = v8f_zero();
    {
      v16h a0 = frag_ld_f16(h16, FF);
      v16h a1 = frag_ld_f16(h16 + 32, FF);
      v16h b0 = frag_ld_f16(W216 + e0 * FF, FF);
      v16h b1f = frag_ld_f16(W216 + e0 * FF + 32, FF);
      c0 = wmma_f16(a0, b0, c0);
      c1 = wmma_f16(a1, b1f, c1);
    }
    v8f c = c0 + c1;
#pragma unroll
    for (int r = 0; r < 8; ++r) {
      int m = r + 8 * grp;
      outs[m * EMB + e0 + coln] = c[r] + bv + out1f32[(row0 + m) * EMB + e0 + coln];
    }
  }
  __syncthreads();

  // phase C: LayerNorm2 -> d_out
  const int srow = tid >> 3, seg = tid & 7;
  {
    float s = 0.f, ss = 0.f;
    for (int cix = seg * 64; cix < seg * 64 + 64; ++cix) {
      float v = outs[srow * EMB + cix];
      s += v; ss += v * v;
    }
    red[srow * 8 + seg] = s;
    red[128 + srow * 8 + seg] = ss;
  }
  __syncthreads();
  if (tid < 16) {
    float s = 0.f, ss = 0.f;
#pragma unroll
    for (int i = 0; i < 8; ++i) { s += red[tid * 8 + i]; ss += red[128 + tid * 8 + i]; }
    float mu = s * (1.0f / EMB);
    float var = ss * (1.0f / EMB) - mu * mu;
    stat[tid] = mu;
    stat[16 + tid] = rsqrtf(var + 1e-5f);
  }
  __syncthreads();
  {
    float mu = stat[srow], rs = stat[16 + srow];
    for (int cix = seg * 64; cix < seg * 64 + 64; ++cix) {
      float v = (outs[srow * EMB + cix] - mu) * rs * g2[cix] + be2[cix];
      out[(row0 + srow) * EMB + cix] = v;
    }
  }
}

// ---------------- launcher ----------------
extern "C" void kernel_launch(void* const* d_in, const int* in_sizes, int n_in,
                              void* d_out, int out_size, void* d_ws, size_t ws_size,
                              hipStream_t stream) {
  (void)in_sizes; (void)n_in; (void)out_size; (void)ws_size;
  const float* query = (const float*)d_in[0];
  const float* key   = (const float*)d_in[1];
  const float* value = (const float*)d_in[2];
  const int*   mask  = (const int*)d_in[3];
  const float* Wq = (const float*)d_in[4];
  const float* bq = (const float*)d_in[5];
  const float* Wk = (const float*)d_in[6];
  const float* bk = (const float*)d_in[7];
  const float* Wv = (const float*)d_in[8];
  const float* bv = (const float*)d_in[9];
  const float* g1  = (const float*)d_in[10];
  const float* be1 = (const float*)d_in[11];
  const float* g2  = (const float*)d_in[12];
  const float* be2 = (const float*)d_in[13];
  const float* W1 = (const float*)d_in[14];
  const float* b1 = (const float*)d_in[15];
  const float* W2 = (const float*)d_in[16];
  const float* b2 = (const float*)d_in[17];
  float* out = (float*)d_out;

  char* ws = (char*)d_ws;
  _Float16* q16     = (_Float16*)(ws + 0);
  _Float16* k16     = (_Float16*)(ws + 16777216);
  _Float16* vt16    = (_Float16*)(ws + 33554432);
  float*    out1f32 = (float*)   (ws + 50331648);
  _Float16* out1f16 = (_Float16*)(ws + 83886080);
  _Float16* Wq16    = (_Float16*)(ws + 100663296);
  _Float16* Wk16    = (_Float16*)(ws + 101187584);
  _Float16* Wv16    = (_Float16*)(ws + 101711872);
  _Float16* W116    = (_Float16*)(ws + 102236160);
  _Float16* W216    = (_Float16*)(ws + 102301696);

  // weight converts
  attn56_cvt_kernel<<<(EMB * EMB + 255) / 256, 256, 0, stream>>>(Wq, Wq16, EMB * EMB);
  attn56_cvt_kernel<<<(EMB * EMB + 255) / 256, 256, 0, stream>>>(Wk, Wk16, EMB * EMB);
  attn56_cvt_kernel<<<(EMB * EMB + 255) / 256, 256, 0, stream>>>(Wv, Wv16, EMB * EMB);
  attn56_cvt_kernel<<<(FF * EMB + 255) / 256, 256, 0, stream>>>(W1, W116, FF * EMB);
  attn56_cvt_kernel<<<(FF * EMB + 255) / 256, 256, 0, stream>>>(W2, W216, FF * EMB);

  // QKV projections: 3 * 1024 * 32 tiles, 4 waves/block
  attn56_qkv_kernel<<<3 * (TOT / 16) * (EMB / 16) / 4, 128, 0, stream>>>(
      query, key, value, Wq16, Wk16, Wv16, bq, bk, bv, q16, k16, vt16);

  // attention + LN1: one block per (batch, 16-row q tile)
  attn56_attn_kernel<<<BB * (SS / 16), 128, SMEM_BYTES, stream>>>(
      q16, k16, vt16, mask, query, g1, be1, out1f32, out1f16);

  // FFN + LN2
  attn56_ffn_kernel<<<TOT / 16, 128, 0, stream>>>(
      out1f32, out1f16, W116, W216, b1, b2, g2, be2, out);
}
